// Model_60249801228370
// MI455X (gfx1250) — compile-verified, matrix operand order checked by reading, not run on hardware
//
#include <hip/hip_runtime.h>
#include <hip/hip_bf16.h>
#include <math.h>

// ---------------------------------------------------------------------------
// Types / WMMA helpers (CDNA5 gfx1250, wave32)
// ---------------------------------------------------------------------------
typedef _Float16 v16h __attribute__((ext_vector_type(16)));
typedef _Float16 h8   __attribute__((ext_vector_type(8)));
typedef float    v8f  __attribute__((ext_vector_type(8)));

// Fragment loader for 16x16x32 f16 WMMA.
// Matrix stored row-major [16 rows][K], 16B-aligned rows.
// Caller passes:  row_base_ptr = mat + (row16 + (lane&15))*ld + ktile*32 + ((lane>>4)*8)
// lane<16  -> k = {0..7, 16..23} (+ktile*32)    lane>=16 -> k = {8..15, 24..31}
__device__ __forceinline__ v16h frag_ld(const _Float16* p) {
  union { v16h v; h8 h[2]; } u;
  u.h[0] = *(const h8*)(p);
  u.h[1] = *(const h8*)(p + 16);
  return u.v;
}

__device__ __forceinline__ v8f wmma_f16(v16h a, v16h b, v8f c) {
  return __builtin_amdgcn_wmma_f32_16x16x32_f16(false, a, false, b,
                                                (short)0, c, false, false);
}

// Branch-free erf (Abramowitz-Stegun 7.1.26, max abs err 1.5e-7).
__device__ __forceinline__ float erf_apx(float x) {
  float ax = fabsf(x);
  float t = 1.0f / (1.0f + 0.3275911f * ax);
  float y = t * (0.254829592f +
            t * (-0.284496736f +
            t * (1.421413741f +
            t * (-1.453152027f +
            t * 1.061405429f))));
  float r = 1.0f - y * __expf(-ax * ax);
  return copysignf(r, x);
}

__device__ __forceinline__ float gelu_exact(float x) {
  return 0.5f * x * (1.0f + erf_apx(x * 0.70710678f));
}

// ---------------------------------------------------------------------------
// Problem constants
// ---------------------------------------------------------------------------
#define NB   16
#define LL   512
#define NV   321
#define DD   128
#define PLEN 16
#define STR  8
#define PP   64          // patches per series
#define NE   8
#define HH   512
#define PRED 96
#define NS   (NB * NV)          // 5136 series
#define NT   ((long long)NS*PP) // 328704 tokens
#define KHEAD (DD * PP)         // 8192

// ---------------------------------------------------------------------------
// 1) Weight convert: f32 -> f16, transposed so B-fragments load contiguously.
//    W1t[e][n(512)][k(128)], W2t[e][n(128)][k(512)], Wht[n(96)][k(8192)]
// ---------------------------------------------------------------------------
__global__ __launch_bounds__(256) void convw_kernel(
    const float* __restrict__ W1, const float* __restrict__ W2,
    const float* __restrict__ Wh,
    _Float16* __restrict__ W1t, _Float16* __restrict__ W2t,
    _Float16* __restrict__ Wht)
{
  const long long N1 = (long long)NE * DD * HH;   // 524288
  const long long N2 = (long long)NE * HH * DD;   // 524288
  const long long N3 = (long long)KHEAD * PRED;   // 786432
  long long i = (long long)blockIdx.x * 256 + threadIdx.x;
  if (i < N1) {
    int e = (int)(i / (DD * HH)); int r = (int)(i % (DD * HH));
    int k = r / HH, n = r % HH;
    W1t[(long long)e * HH * DD + (long long)n * DD + k] = (_Float16)W1[i];
  } else if (i < N1 + N2) {
    long long j = i - N1;
    int e = (int)(j / (HH * DD)); int r = (int)(j % (HH * DD));
    int k = r / DD, n = r % DD;
    W2t[(long long)e * HH * DD + (long long)n * HH + k] = (_Float16)W2[j];
  } else if (i < N1 + N2 + N3) {
    long long j = i - N1 - N2;
    int k = (int)(j / PRED), n = (int)(j % PRED);
    Wht[(long long)n * KHEAD + k] = (_Float16)Wh[j];
  }
}

// ---------------------------------------------------------------------------
// 2) Per-series mean / stdev over L
// ---------------------------------------------------------------------------
__global__ __launch_bounds__(256) void stats_kernel(
    const float* __restrict__ x, float* __restrict__ means,
    float* __restrict__ stdev)
{
  __shared__ float s1[256];
  __shared__ float s2[256];
  int s = blockIdx.x;
  int b = s / NV, nv = s % NV;
  int tid = threadIdx.x;
  float a = x[((long long)b * LL + tid) * NV + nv];
  float c = x[((long long)b * LL + tid + 256) * NV + nv];
  s1[tid] = a + c;
  s2[tid] = a * a + c * c;
  __syncthreads();
  for (int o = 128; o > 0; o >>= 1) {
    if (tid < o) { s1[tid] += s1[tid + o]; s2[tid] += s2[tid + o]; }
    __syncthreads();
  }
  if (tid == 0) {
    float m = s1[0] / (float)LL;
    float v = s2[0] / (float)LL - m * m;
    means[s] = m;
    stdev[s] = sqrtf(v + 1e-5f);
  }
}

// ---------------------------------------------------------------------------
// 3) Patch embed + sinusoidal PE + softmax gating + top-2 + aux accumulators.
//    One 128-thread block per token.
// ---------------------------------------------------------------------------
__global__ __launch_bounds__(128) void embed_gate_kernel(
    const float* __restrict__ x_enc, const float* __restrict__ Wval,
    const float* __restrict__ Wg, const float* __restrict__ means,
    const float* __restrict__ stdev, _Float16* __restrict__ t16,
    float* __restrict__ gbuf, float* __restrict__ accsl)
{
  __shared__ float sx[PLEN];
  __shared__ float tv[DD];
  __shared__ float lg[NE];
  long long n = blockIdx.x;
  int s = (int)(n >> 6);   // / PP
  int p = (int)(n & 63);
  int b = s / NV, nv = s % NV;
  int tid = threadIdx.x;

  if (tid < PLEN) {
    int l = p * STR + tid; if (l > LL - 1) l = LL - 1;   // edge pad
    sx[tid] = (x_enc[((long long)b * LL + l) * NV + nv] - means[s]) / stdev[s];
  }
  __syncthreads();

  int d = tid;
  int i2 = d & ~1;
  float div = __expf(-(float)i2 * 0.0719557842f);  // ln(10000)/128
  float ang = (float)p * div;
  float acc = (d & 1) ? __cosf(ang) : __sinf(ang);
#pragma unroll
  for (int j = 0; j < PLEN; ++j) acc += sx[j] * Wval[j * DD + d];
  t16[n * DD + d] = (_Float16)acc;
  tv[d] = acc;
  __syncthreads();

  if (tid < NE) {
    float l = 0.f;
    for (int k = 0; k < DD; ++k) l += tv[k] * Wg[k * NE + tid];
    lg[tid] = l;
  }
  __syncthreads();

  if (tid == 0) {
    float mx = lg[0];
    for (int e = 1; e < NE; ++e) mx = fmaxf(mx, lg[e]);
    float ex[NE], ssum = 0.f;
    for (int e = 0; e < NE; ++e) { ex[e] = __expf(lg[e] - mx); ssum += ex[e]; }
    float lse = mx + __logf(ssum);
    float pr[NE];
    for (int e = 0; e < NE; ++e) pr[e] = ex[e] / ssum;
    int i1 = 0;
    for (int e = 1; e < NE; ++e) if (pr[e] > pr[i1]) i1 = e;   // first-index ties
    int j2 = -1;
    for (int e = 0; e < NE; ++e)
      if (e != i1 && (j2 < 0 || pr[e] > pr[j2])) j2 = e;
    int slot = ((int)(n & 255)) * 20;
    for (int e = 0; e < NE; ++e) {
      float g = (e == i1) ? pr[i1] : ((e == j2) ? pr[j2] : 0.f);
      gbuf[n * NE + e] = g;
      atomicAdd(&accsl[slot + e], pr[e]);
    }
    atomicAdd(&accsl[slot + 8 + i1], 1.f);
    atomicAdd(&accsl[slot + 8 + j2], 1.f);
    atomicAdd(&accsl[slot + 16], lse * lse);
  }
}

// ---------------------------------------------------------------------------
// 4) MoE FFN with WMMA. One block (4 waves) per series = 64-token M tile.
//    For each expert: phase1 T(64x128)@W1 -> GELU -> H slab (64x256 f16 LDS),
//    phase2 H@W2 accumulated with per-row gate into f32 VGPR accumulators.
//    H split in two K-halves to fit 50KB static LDS.
// ---------------------------------------------------------------------------
__global__ __launch_bounds__(128) void moe_kernel(
    const _Float16* __restrict__ t16, const float* __restrict__ gbuf,
    const _Float16* __restrict__ W1t, const _Float16* __restrict__ W2t,
    const float* __restrict__ b1, const float* __restrict__ b2,
    float* __restrict__ outb)
{
  __shared__ __align__(16) _Float16 Tt[64 * DD];       // 16 KB
  __shared__ __align__(16) _Float16 Ht[64 * 256];      // 32 KB
  __shared__ float Gt[NE * 64];                        //  2 KB

  const int s = blockIdx.x;
  const int tid = threadIdx.x;
  const int w = tid >> 5;
  const int lane = tid & 31;
  const int ln = lane & 15;
  const int kh = (lane >> 4) * 8;     // also the C-row offset for hi half-wave
  const long long n0 = (long long)s * PP;

  // stage token tile (coalesced 16B vectors) and gates
  {
    const h8* src = (const h8*)(t16 + n0 * DD);
    h8* dst = (h8*)Tt;
    for (int i = tid; i < 64 * DD / 8; i += 128) dst[i] = src[i];
    for (int i = tid; i < NE * 64; i += 128) {
      int e = i >> 6, p = i & 63;
      Gt[i] = gbuf[(n0 + p) * NE + e];
    }
  }

  v8f outacc[8];
#pragma unroll
  for (int dt = 0; dt < 8; ++dt)
#pragma unroll
    for (int i = 0; i < 8; ++i) outacc[dt][i] = 0.f;

  for (int e = 0; e < NE; ++e) {
    const _Float16* W1e = W1t + (long long)e * HH * DD;   // [n512][k128]
    const _Float16* W2e = W2t + (long long)e * DD * HH;   // [n128][k512]

    // prefetch a slice of the *next* expert's weights (global_prefetch_b8)
    {
      int en = (e + 1) & 7;
      const _Float16* nW1 = W1t + (long long)en * HH * DD;
      const _Float16* nW2 = W2t + (long long)en * DD * HH;
#pragma unroll
      for (int j = 0; j < 4; ++j) {
        __builtin_prefetch(nW1 + ((long long)tid + j * 128) * 32, 0, 1);
        __builtin_prefetch(nW2 + ((long long)tid + j * 128) * 32, 0, 1);
      }
    }

    for (int half = 0; half < 2; ++half) {
      __syncthreads();   // Ht reuse fence (also covers initial Tt staging)
      // ---- phase 1: wave w computes H columns [half*256 + w*64 .. +64) ----
      for (int nt4 = 0; nt4 < 4; ++nt4) {
        int ntloc = w * 4 + nt4;            // local H col tile (0..15)
        int gc = half * 256 + ntloc * 16;   // global W1 column base
        v8f acc[4];
#pragma unroll
        for (int mt = 0; mt < 4; ++mt)
#pragma unroll
          for (int i = 0; i < 8; ++i) acc[mt][i] = 0.f;
#pragma unroll
        for (int kt = 0; kt < 4; ++kt) {
          v16h bf = frag_ld(W1e + (gc + ln) * DD + kt * 32 + kh);
#pragma unroll
          for (int mt = 0; mt < 4; ++mt) {
            v16h af = frag_ld(Tt + (mt * 16 + ln) * DD + kt * 32 + kh);
            acc[mt] = wmma_f16(af, bf, acc[mt]);
          }
        }
        float bb = b1[e * HH + gc + ln];
#pragma unroll
        for (int mt = 0; mt < 4; ++mt)
#pragma unroll
          for (int i = 0; i < 8; ++i) {
            int m = mt * 16 + i + kh;
            float gl = gelu_exact(acc[mt][i] + bb);
            Ht[m * 256 + ntloc * 16 + ln] = (_Float16)gl;
          }
      }
      __syncthreads();
      // ---- phase 2: wave w owns rows [w*16 .. w*16+16), all 8 D tiles ----
      for (int dt = 0; dt < 8; ++dt) {
        v8f accA, accB;
#pragma unroll
        for (int i = 0; i < 8; ++i) { accA[i] = 0.f; accB[i] = 0.f; }
#pragma unroll
        for (int kt = 0; kt < 8; kt += 2) {
          v16h af0 = frag_ld(Ht + (w * 16 + ln) * 256 + kt * 32 + kh);
          v16h bf0 = frag_ld(W2e + (dt * 16 + ln) * HH + half * 256 + kt * 32 + kh);
          v16h af1 = frag_ld(Ht + (w * 16 + ln) * 256 + (kt + 1) * 32 + kh);
          v16h bf1 = frag_ld(W2e + (dt * 16 + ln) * HH + half * 256 + (kt + 1) * 32 + kh);
          accA = wmma_f16(af0, bf0, accA);
          accB = wmma_f16(af1, bf1, accB);
        }
#pragma unroll
        for (int i = 0; i < 8; ++i) {
          int m = w * 16 + i + kh;
          float add = accA[i] + accB[i];
          if (half) add += b2[e * DD + dt * 16 + ln];
          outacc[dt][i] += Gt[e * 64 + m] * add;
        }
      }
    }
  }

#pragma unroll
  for (int dt = 0; dt < 8; ++dt)
#pragma unroll
    for (int i = 0; i < 8; ++i) {
      int m = w * 16 + i + kh;
      outb[(n0 + m) * DD + dt * 16 + ln] = outacc[dt][i];
    }
}

// ---------------------------------------------------------------------------
// 5) Transpose enc (per series: [p][d] -> k = d*64+p) to f16 for the head GEMM.
//    Bank-conflict-free via 129-float padded LDS rows.
// ---------------------------------------------------------------------------
__global__ __launch_bounds__(128) void transp_kernel(
    const float* __restrict__ outb, _Float16* __restrict__ Ah)
{
  __shared__ float ldsT[64 * 129];
  int s = blockIdx.x;
  int tid = threadIdx.x;
  const float* src = outb + (long long)s * KHEAD;
  for (int i = tid; i < KHEAD; i += 128) {
    int p = i >> 7, d = i & 127;
    ldsT[p * 129 + d] = src[i];
  }
  __syncthreads();
  _Float16* dst = Ah + (long long)s * KHEAD;
  for (int i = tid; i < KHEAD; i += 128) {
    int d = i >> 6, p = i & 63;
    dst[i] = (_Float16)ldsT[p * 129 + d];
  }
}

// ---------------------------------------------------------------------------
// 6) Head GEMM (5136 x 8192) @ (8192 x 96) with WMMA + bias + de-normalize
//    + scatter to (B, PRED, NV) output layout. One wave per 16x16 tile.
//    4 interleaved accumulators break the WMMA dependency chain.
// ---------------------------------------------------------------------------
__global__ __launch_bounds__(32) void head_kernel(
    const _Float16* __restrict__ Ah, const _Float16* __restrict__ Wht,
    const float* __restrict__ bh, const float* __restrict__ means,
    const float* __restrict__ stdev, float* __restrict__ dout)
{
  int st = blockIdx.x;    // series tile 0..320
  int nt = blockIdx.y;    // output tile 0..5
  int lane = threadIdx.x;
  int ln = lane & 15;
  int kh = (lane >> 4) * 8;

  v8f a0, a1, a2, a3;
#pragma unroll
  for (int i = 0; i < 8; ++i) { a0[i] = 0.f; a1[i] = 0.f; a2[i] = 0.f; a3[i] = 0.f; }

  const _Float16* arow = Ah  + (long long)(st * 16 + ln) * KHEAD + kh;
  const _Float16* brow = Wht + (long long)(nt * 16 + ln) * KHEAD + kh;
  for (int kt = 0; kt < KHEAD / 32; kt += 4) {
    a0 = wmma_f16(frag_ld(arow + (kt + 0) * 32), frag_ld(brow + (kt + 0) * 32), a0);
    a1 = wmma_f16(frag_ld(arow + (kt + 1) * 32), frag_ld(brow + (kt + 1) * 32), a1);
    a2 = wmma_f16(frag_ld(arow + (kt + 2) * 32), frag_ld(brow + (kt + 2) * 32), a2);
    a3 = wmma_f16(frag_ld(arow + (kt + 3) * 32), frag_ld(brow + (kt + 3) * 32), a3);
  }
  v8f acc = (a0 + a1) + (a2 + a3);

#pragma unroll
  for (int i = 0; i < 8; ++i) {
    int srow = st * 16 + i + kh;
    int pr = nt * 16 + ln;
    int b = srow / NV, nv = srow % NV;
    float val = (acc[i] + bh[pr]) * stdev[srow] + means[srow];
    dout[((long long)b * PRED + pr) * NV + nv] = val;
  }
}

// ---------------------------------------------------------------------------
// 7) Aux losses reduction
// ---------------------------------------------------------------------------
__global__ void aux_kernel(const float* __restrict__ accsl,
                           float* __restrict__ dout)
{
  if (threadIdx.x == 0 && blockIdx.x == 0) {
    float Sp[NE], Sc[NE], Sz = 0.f;
    for (int e = 0; e < NE; ++e) { Sp[e] = 0.f; Sc[e] = 0.f; }
    for (int sl = 0; sl < 256; ++sl) {
      for (int e = 0; e < NE; ++e) {
        Sp[e] += accsl[sl * 20 + e];
        Sc[e] += accsl[sl * 20 + 8 + e];
      }
      Sz += accsl[sl * 20 + 16];
    }
    const float N = (float)NT;
    float bal = 0.f;
    for (int e = 0; e < NE; ++e) bal += (Sp[e] / N) * (Sc[e] / (N * 2.f));
    bal *= (float)NE;
    float z = Sz / N;
    dout[(long long)NB * PRED * NV] = 0.01f * bal + 0.001f * z;
  }
}

// ---------------------------------------------------------------------------
// Host launcher
// ---------------------------------------------------------------------------
extern "C" void kernel_launch(void* const* d_in, const int* in_sizes, int n_in,
                              void* d_out, int out_size, void* d_ws, size_t ws_size,
                              hipStream_t stream) {
  (void)in_sizes; (void)n_in; (void)out_size; (void)ws_size;
  const float* x_enc  = (const float*)d_in[0];
  const float* W_val  = (const float*)d_in[4];
  const float* W_g    = (const float*)d_in[5];
  const float* W1     = (const float*)d_in[6];
  const float* b1     = (const float*)d_in[7];
  const float* W2     = (const float*)d_in[8];
  const float* b2     = (const float*)d_in[9];
  const float* W_head = (const float*)d_in[10];
  const float* b_head = (const float*)d_in[11];
  float* out = (float*)d_out;

  char* ws = (char*)d_ws;
  size_t off = 0;
  auto alloc = [&](size_t bytes) -> void* {
    void* p = ws + off;
    off += (bytes + 255) & ~(size_t)255;
    return p;
  };
  float*    means = (float*)alloc((size_t)NS * 4);
  float*    stdev = (float*)alloc((size_t)NS * 4);
  float*    accsl = (float*)alloc((size_t)256 * 20 * 4);
  float*    gbuf  = (float*)alloc((size_t)NT * NE * 4);
  _Float16* t16   = (_Float16*)alloc((size_t)NT * DD * 2);
  float*    outb  = (float*)alloc((size_t)NT * DD * 4);
  _Float16* W1t   = (_Float16*)alloc((size_t)NE * DD * HH * 2);
  _Float16* W2t   = (_Float16*)alloc((size_t)NE * HH * DD * 2);
  _Float16* Wht   = (_Float16*)alloc((size_t)KHEAD * PRED * 2);
  _Float16* Ah    = t16;   // alias: token f16 buffer dead after moe_kernel

  hipMemsetAsync(accsl, 0, 256 * 20 * 4, stream);
  convw_kernel<<<7168, 256, 0, stream>>>(W1, W2, W_head, W1t, W2t, Wht);
  stats_kernel<<<NS, 256, 0, stream>>>(x_enc, means, stdev);
  embed_gate_kernel<<<(unsigned)NT, 128, 0, stream>>>(x_enc, W_val, W_g, means,
                                                      stdev, t16, gbuf, accsl);
  moe_kernel<<<NS, 128, 0, stream>>>(t16, gbuf, W1t, W2t, b1, b2, outb);
  transp_kernel<<<NS, 128, 0, stream>>>(outb, Ah);
  head_kernel<<<dim3(NS / 16, PRED / 16), 32, 0, stream>>>(Ah, Wht, b_head,
                                                           means, stdev, out);
  aux_kernel<<<1, 32, 0, stream>>>(accsl, out);
}
